// DTTreeGRU_90108413870167
// MI455X (gfx1250) — matrix operand order repaired and checked by reading, not compiled, not run on hardware
//
#include <hip/hip_runtime.h>
#include <hip/hip_bf16.h>
#include <math.h>

// Problem constants (from reference): L=1024, B=64, D=256, H=256, K=4
#define NL      1024
#define NB      64
#define ND      256
#define NH      256
#define TM      16          // rows (node-batch pairs) per block
#define KG      768         // gate GEMM K  (= D + H + H)
#define NG      1280        // gate GEMM N  (= 5*H)

typedef __bf16 bf16_t;
typedef __attribute__((ext_vector_type(16))) __bf16 v16bf;
typedef __attribute__((ext_vector_type(8)))  __bf16 v8bf;
typedef __attribute__((ext_vector_type(8)))  float  v8f;
typedef __attribute__((ext_vector_type(4)))  unsigned int u32x4;
typedef __attribute__((ext_vector_type(8)))  int          i32x8;
typedef __attribute__((ext_vector_type(4)))  int          i32x4;

// ---------------------------------------------------------------------------
// Fragment loaders (WMMA 16x16x32 bf16, wave32; this file's contiguous k order, same for A and B)
// A (16xK region in LDS, row-major, row stride = `stride` elems):
//   lane l: M = l&15, half = l>>4; elems 0..15 = K[k0+16h .. k0+16h+15]
__device__ __forceinline__ v16bf load_a_frag(const bf16_t* base, int stride, int k0) {
    const int l    = threadIdx.x & 31;
    const int m    = l & 15;
    const int half = l >> 4;
    const bf16_t* p = base + m * stride + k0 + half * 16;
    v8bf lo = *(const v8bf*)(p);
    v8bf hi = *(const v8bf*)(p + 8);
    v16bf a;
#pragma unroll
    for (int i = 0; i < 8; ++i) { a[i] = lo[i]; a[8 + i] = hi[i]; }
    return a;
}

// B from N-major weight storage W[n][k] (ldk = K stride):
//   lane l: N = l&15, half = l>>4; elems = K[k0+16h .. k0+16h+15]
__device__ __forceinline__ v16bf load_b_frag(const bf16_t* w_nk, int n0, int ldk, int k0) {
    const int l    = threadIdx.x & 31;
    const int n    = l & 15;
    const int half = l >> 4;
    return *(const v16bf*)(w_nk + (size_t)(n0 + n) * ldk + k0 + half * 16);
}

__device__ __forceinline__ v8f wmma_bf16(v16bf a, v16bf b, v8f c) {
    return __builtin_amdgcn_wmma_f32_16x16x32_bf16(false, a, false, b, (short)0, c,
                                                   false, false);
}

// ---------------------------------------------------------------------------
// Fused per-level kernel: gates GEMM + sigmoid + cell GEMM + tanh + combine
__global__ __launch_bounds__(256) void tgru_level_kernel(
    int node_start,
    const bf16_t* __restrict__ x_bf,     // [NL][NB][ND] bf16
    bf16_t*       __restrict__ h_bf,     // [NL+1][NB][NH] bf16
    float*        __restrict__ h_f32,    // [NL+1][NB][NH] f32
    const bf16_t* __restrict__ Wg,       // [NG][KG] bf16 (gih|glh|grh)
    const bf16_t* __restrict__ Wc,       // [NH][KG] bf16 (cih|clh|crh)
    const float*  __restrict__ b_gih,    // [NG]
    const float*  __restrict__ b_cih)    // [NH]
{
    __shared__ bf16_t Xs [TM * ND];      //  8 KB : x rows
    __shared__ bf16_t LHs[TM * NH];      //  8 KB : left child-sum h
    __shared__ bf16_t RHs[TM * NH];      //  8 KB : right child-sum h
    __shared__ bf16_t A2s[TM * 512];     // 16 KB : [rl*lh | rr*rh]
    __shared__ bf16_t Zs [TM * 768];     // 24 KB : [zl | zr | z]

    const int node = node_start + (blockIdx.x >> 2);
    const int row0 = (blockIdx.x & 3) * TM;          // batch offset of this tile
    const int tid  = threadIdx.x;
    const int wave = tid >> 5;
    const int lane = tid & 31;

    // ---- stage X tile (16 rows x 256, contiguous 8KB) via Tensor Data Mover ----
    const bf16_t* xsrc = x_bf + ((size_t)node * NB + row0) * ND;
#if __has_builtin(__builtin_amdgcn_tensor_load_to_lds) && __has_builtin(__builtin_amdgcn_s_wait_tensorcnt)
    if (wave == 0) {
        unsigned long long ga = (unsigned long long)(uintptr_t)xsrc;
        unsigned int lds_off  = (unsigned int)(uintptr_t)(&Xs[0]);  // low 32b = LDS offset
        u32x4 g0;
        g0[0] = 1u;                                   // count=1 valid descriptor
        g0[1] = lds_off;                              // lds_addr
        g0[2] = (unsigned int)ga;                     // global_addr[31:0]
        g0[3] = (unsigned int)(ga >> 32) | (2u << 30);// global_addr[56:32] | type=2
        i32x8 g1;
        g1[0] = 0x00010000;                           // data_size=1 (2 bytes)
        g1[1] = (int)(4096u << 16);                   // tensor_dim0 = 4096 (1-D run)
        g1[2] = (int)(1u << 16);                      // tensor_dim1 = 1
        g1[3] = (int)(4096u << 16);                   // tile_dim0 = 4096
        g1[4] = 1;                                    // tile_dim1 = 1
        g1[5] = 4096;                                 // tensor_dim0_stride = 4096
        g1[6] = 0; g1[7] = 0;
        i32x4 gz;  gz[0] = 0; gz[1] = 0; gz[2] = 0; gz[3] = 0;
        i32x8 gz8; gz8[0] = 0; gz8[1] = 0; gz8[2] = 0; gz8[3] = 0;
                   gz8[4] = 0; gz8[5] = 0; gz8[6] = 0; gz8[7] = 0;
        __builtin_amdgcn_tensor_load_to_lds(g0, g1, gz, gz, gz8, 0);
        __builtin_amdgcn_s_wait_tensorcnt(0);
    }
#else
    for (int e = tid * 8; e < TM * ND; e += 256 * 8)
        *(v8bf*)(Xs + e) = *(const v8bf*)(xsrc + e);
#endif

    // ---- stage child hidden sums: LH = h[c0]+h[c1], RH = h[c2]+h[c3] ----
    {
        const int cb = 4 * node + 1;
        const int c0 = (cb + 0 < NL) ? cb + 0 : NL;
        const int c1 = (cb + 1 < NL) ? cb + 1 : NL;
        const int c2 = (cb + 2 < NL) ? cb + 2 : NL;
        const int c3 = (cb + 3 < NL) ? cb + 3 : NL;
        const bf16_t* hp0 = h_bf + ((size_t)c0 * NB + row0) * NH;
        const bf16_t* hp1 = h_bf + ((size_t)c1 * NB + row0) * NH;
        const bf16_t* hp2 = h_bf + ((size_t)c2 * NB + row0) * NH;
        const bf16_t* hp3 = h_bf + ((size_t)c3 * NB + row0) * NH;
        for (int e = tid * 8; e < TM * NH; e += 256 * 8) {
            v8bf a0 = *(const v8bf*)(hp0 + e);
            v8bf a1 = *(const v8bf*)(hp1 + e);
            v8bf b0 = *(const v8bf*)(hp2 + e);
            v8bf b1 = *(const v8bf*)(hp3 + e);
            v8bf lh, rh;
#pragma unroll
            for (int i = 0; i < 8; ++i) {
                lh[i] = (bf16_t)((float)a0[i] + (float)a1[i]);
                rh[i] = (bf16_t)((float)b0[i] + (float)b1[i]);
            }
            *(v8bf*)(LHs + e) = lh;
            *(v8bf*)(RHs + e) = rh;
        }
    }
    __syncthreads();

    // ---- phase 1: gates GEMM (16 x 1280, K=768) ----
    // 80 column tiles of 16 over 8 waves; A = [X | LH | RH] from LDS.
    for (int t = wave; t < NG / 16; t += 8) {
        v8f acc = {};
        for (int ki = 0; ki < KG / 32; ++ki) {
            const int k0 = ki * 32;
            v16bf a = (k0 < 256) ? load_a_frag(Xs,  ND, k0)
                    : (k0 < 512) ? load_a_frag(LHs, NH, k0 - 256)
                                 : load_a_frag(RHs, NH, k0 - 512);
            v16bf b = load_b_frag(Wg, t * 16, KG, k0);
            acc = wmma_bf16(a, b, acc);
        }
        const int colg = t * 16 + (lane & 15);
        const float bias = b_gih[colg];
        const int rbase  = (lane >> 4) ? 8 : 0;
#pragma unroll
        for (int j = 0; j < 8; ++j) {
            const int r   = rbase + j;
            const float s = 1.0f / (1.0f + __expf(-(acc[j] + bias)));
            if (colg < 256) {                 // rl -> rl * lh into A2 K-slot
                A2s[r * 512 + colg] = (bf16_t)(s * (float)LHs[r * NH + colg]);
            } else if (colg < 512) {          // rr -> rr * rh into A2 K-slot
                A2s[r * 512 + colg] = (bf16_t)(s * (float)RHs[r * NH + (colg - 256)]);
            } else {                          // zl | zr | z
                Zs[r * 768 + (colg - 512)] = (bf16_t)s;
            }
        }
    }
    __syncthreads();

    // ---- phase 2: cell GEMM (16 x 256, K=768 over [X | A2]) + combine ----
    for (int t = wave; t < NH / 16; t += 8) {
        v8f acc = {};
        for (int ki = 0; ki < KG / 32; ++ki) {
            const int k0 = ki * 32;
            v16bf a = (k0 < 256) ? load_a_frag(Xs, ND, k0)
                                 : load_a_frag(A2s, 512, k0 - 256);
            v16bf b = load_b_frag(Wc, t * 16, KG, k0);
            acc = wmma_bf16(a, b, acc);
        }
        const int col   = t * 16 + (lane & 15);
        const float bias = b_cih[col];
        const int rbase  = (lane >> 4) ? 8 : 0;
#pragma unroll
        for (int j = 0; j < 8; ++j) {
            const int r = rbase + j;
            const float cell = tanhf(acc[j] + bias);
            const float lh = (float)LHs[r * NH + col];
            const float rh = (float)RHs[r * NH + col];
            const float zl = (float)Zs[r * 768 + col];
            const float zr = (float)Zs[r * 768 + 256 + col];
            const float zz = (float)Zs[r * 768 + 512 + col];
            const float hid = zl * lh + zr * rh + zz * cell;
            const size_t off = ((size_t)node * NB + row0 + r) * NH + col;
            h_f32[off] = hid;
            h_bf [off] = (bf16_t)hid;
        }
    }
}

// ---------------------------------------------------------------------------
// Prep / epilogue kernels
__global__ void cvt_to_bf16_kernel(const float* __restrict__ src,
                                   bf16_t* __restrict__ dst, int n) {
    int i = blockIdx.x * blockDim.x + threadIdx.x;
    if (i < n) dst[i] = (bf16_t)src[i];
}

// Wg[n][k] = k<256 ? W_gih[n][k] : k<512 ? W_glh[n][k-256] : W_grh[n][k-512]
__global__ void pack_w_kernel(const float* __restrict__ Wi,
                              const float* __restrict__ Wl,
                              const float* __restrict__ Wr,
                              bf16_t* __restrict__ Wp, int nrows) {
    int i = blockIdx.x * blockDim.x + threadIdx.x;
    if (i >= nrows * KG) return;
    int n = i / KG, k = i % KG;
    float v = (k < 256) ? Wi[n * 256 + k]
            : (k < 512) ? Wl[n * 256 + (k - 256)]
                        : Wr[n * 256 + (k - 512)];
    Wp[i] = (bf16_t)v;
}

__global__ void zero_pad_kernel(bf16_t* __restrict__ h_bf,
                                float* __restrict__ h_f32) {
    int i = blockIdx.x * blockDim.x + threadIdx.x;
    if (i < NB * NH) {
        size_t off = (size_t)NL * NB * NH + i;   // node index NL == zero node
        h_f32[off] = 0.0f;
        h_bf [off] = (bf16_t)0.0f;
    }
}

// outputs = concat(hn[:,1:], hn[:,:1], axis=1)  (B,L,H), then root hn[:,0] (B,H)
__global__ void emit_out_kernel(const float* __restrict__ h_f32,
                                float* __restrict__ out) {
    const int total = NB * NL * NH + NB * NH;
    int i = blockIdx.x * blockDim.x + threadIdx.x;
    if (i >= total) return;
    if (i < NB * NL * NH) {
        int c = i & (NH - 1);
        int l = (i >> 8) & (NL - 1);
        int b = i >> 18;
        int src = (l + 1) & (NL - 1);
        out[i] = h_f32[((size_t)src * NB + b) * NH + c];
    } else {
        int r = i - NB * NL * NH;
        int c = r & (NH - 1);
        int b = r >> 8;
        out[i] = h_f32[(size_t)b * NH + c];      // node 0
    }
}

// ---------------------------------------------------------------------------
extern "C" void kernel_launch(void* const* d_in, const int* in_sizes, int n_in,
                              void* d_out, int out_size, void* d_ws, size_t ws_size,
                              hipStream_t stream) {
    const float* inputs = (const float*)d_in[0];   // (L,B,D)
    const float* W_gih  = (const float*)d_in[3];   // (1280,256)
    const float* b_gih  = (const float*)d_in[4];   // (1280,)
    const float* W_glh  = (const float*)d_in[5];   // (1280,256)
    const float* W_grh  = (const float*)d_in[6];   // (1280,256)
    const float* W_cih  = (const float*)d_in[7];   // (256,256)
    const float* b_cih  = (const float*)d_in[8];   // (256,)
    const float* W_clh  = (const float*)d_in[9];   // (256,256)
    const float* W_crh  = (const float*)d_in[10];  // (256,256)
    (void)in_sizes; (void)n_in; (void)out_size; (void)ws_size;

    // workspace layout
    char* ws = (char*)d_ws;
    const size_t xcnt = (size_t)NL * NB * ND;            // 16.7M
    const size_t hcnt = (size_t)(NL + 1) * NB * NH;      // 16.8M
    bf16_t* x_bf  = (bf16_t*)(ws);                       size_t off = xcnt * 2;
    float*  h_f32 = (float*) (ws + off);                 off += hcnt * 4;
    bf16_t* h_bf  = (bf16_t*)(ws + off);                 off += hcnt * 2;
    bf16_t* Wg    = (bf16_t*)(ws + off);                 off += (size_t)NG * KG * 2;
    bf16_t* Wc    = (bf16_t*)(ws + off);

    // ---- prep ----
    cvt_to_bf16_kernel<<<(int)((xcnt + 255) / 256), 256, 0, stream>>>(inputs, x_bf, (int)xcnt);
    pack_w_kernel<<<(NG * KG + 255) / 256, 256, 0, stream>>>(W_gih, W_glh, W_grh, Wg, NG);
    pack_w_kernel<<<(NH * KG + 255) / 256, 256, 0, stream>>>(W_cih, W_clh, W_crh, Wc, NH);
    zero_pad_kernel<<<(NB * NH + 255) / 256, 256, 0, stream>>>(h_bf, h_f32);

    // ---- levels, deepest first (4-ary heap layout, hardcoded level ranges) ----
    static const int lv_start[6] = { 0, 1, 5, 21, 85, 341 };
    static const int lv_count[6] = { 1, 4, 16, 64, 256, 683 };
    for (int d = 5; d >= 0; --d) {
        dim3 grid(lv_count[d] * 4);   // 4 row-tiles of 16 batch rows each
        tgru_level_kernel<<<grid, 256, 0, stream>>>(
            lv_start[d], x_bf, h_bf, h_f32, Wg, Wc, b_gih, b_cih);
    }

    // ---- epilogue: rotate + root ----
    const int total = NB * NL * NH + NB * NH;
    emit_out_kernel<<<(total + 255) / 256, 256, 0, stream>>>(h_f32, (float*)d_out);
}
